// MaskedGraphTransformer_6786048328250
// MI455X (gfx1250) — matrix operand order repaired and checked by reading, not compiled, not run on hardware
//
#include <hip/hip_runtime.h>
#include <hip/hip_bf16.h>
#include <math.h>

#define D_MODEL 256
#define NHEAD   8
#define HEADD   32
#define NLAYER  2
#define FFDIM   1024
#define BATCH   8
#define SEQ     1024
#define NEDGE   16384

typedef __attribute__((ext_vector_type(16))) _Float16 v16h;
typedef __attribute__((ext_vector_type(8)))  float    v8f;

union FragA {
  uint4    q[2];
  v16h     h;
  _Float16 e[16];
};

static __device__ __forceinline__ v8f wmma_f16(const v16h a, const v16h b, const v8f c) {
  // D = A(16x32 f16) * B(32x16 f16) + C(16x16 f32)
  return __builtin_amdgcn_wmma_f32_16x16x32_f16(false, a, false, b, (short)0, c, false, false);
}

static __device__ __forceinline__ void load16h(FragA& f, const float* p) {
  float4 a = *(const float4*)(p);
  float4 b = *(const float4*)(p + 4);
  float4 c = *(const float4*)(p + 8);
  float4 d = *(const float4*)(p + 12);
  f.e[0]  = (_Float16)a.x; f.e[1]  = (_Float16)a.y; f.e[2]  = (_Float16)a.z; f.e[3]  = (_Float16)a.w;
  f.e[4]  = (_Float16)b.x; f.e[5]  = (_Float16)b.y; f.e[6]  = (_Float16)b.z; f.e[7]  = (_Float16)b.w;
  f.e[8]  = (_Float16)c.x; f.e[9]  = (_Float16)c.y; f.e[10] = (_Float16)c.z; f.e[11] = (_Float16)c.w;
  f.e[12] = (_Float16)d.x; f.e[13] = (_Float16)d.y; f.e[14] = (_Float16)d.z; f.e[15] = (_Float16)d.w;
}

static __device__ __forceinline__ float gelu_exact(float x) {
  return 0.5f * x * (1.0f + erff(x * 0.7071067811865476f));
}

// ---------------------------------------------------------------------------
// Bias construction: bias[b][i][j] = -inf; diag = 0; edges get log(m + 1e-9)
// ---------------------------------------------------------------------------
__global__ __launch_bounds__(256) void bias_init_kernel(float* __restrict__ bias) {
  size_t i = (size_t)blockIdx.x * 256 + threadIdx.x;   // < B*N*N (exact)
  int rc = (int)(i & (size_t)(SEQ * SEQ - 1));
  int r = rc >> 10;
  int c = rc & (SEQ - 1);
  bias[i] = (r == c) ? 0.0f : -INFINITY;
}

__global__ __launch_bounds__(256) void edge_scatter_kernel(const int* __restrict__ ei,
                                                           const float* __restrict__ m,
                                                           float* __restrict__ bias) {
  int i = blockIdx.x * 256 + threadIdx.x;              // < B*E (exact)
  int b = i >> 14;
  int e = i & (NEDGE - 1);
  int row = ei[(size_t)b * 2 * NEDGE + e];
  int col = ei[(size_t)b * 2 * NEDGE + NEDGE + e];
  float bv = logf(m[(size_t)b * NEDGE + e] + 1e-9f);
  float* bp = bias + (size_t)b * SEQ * SEQ;
  bp[(size_t)row * SEQ + col] = bv;
  bp[(size_t)col * SEQ + row] = bv;
}

// ---------------------------------------------------------------------------
// GEMM: C[M,Nout] = act(A[M,K] * W[Nout,K]^T + bvec). Block tile 64x64,
// 8 wave32 waves, LDS-staged f16 operands, WMMA f32_16x16x32_f16.
// ACT: 0 = none, 1 = exact GELU.
// ---------------------------------------------------------------------------
template <int ACT>
__global__ __launch_bounds__(256) void gemm_kernel(const float* __restrict__ A,
                                                   const float* __restrict__ W,
                                                   const float* __restrict__ bvec,
                                                   float* __restrict__ C,
                                                   int M, int K, int Nout) {
  __shared__ __align__(16) _Float16 As[64 * 40];   // 64 rows x 32 K, stride 40 halves (80B)
  __shared__ __align__(16) _Float16 Bs[64 * 40];   // 64 cols x 32 K

  const int tid  = threadIdx.x;
  const int lane = tid & 31;
  const int wid  = tid >> 5;
  const int rowBase = blockIdx.y * 64;
  const int colBase = blockIdx.x * 64;

  // staging assignment: thread -> (row, 4-float quad)
  const int sr = tid >> 2;            // 0..63
  const int sq = (tid & 3) * 4;       // 0,4,8,12

  // WMMA fragment addressing (contiguous K: lane half h holds K 16h..16h+15 in A and B)
  const int l15   = lane & 15;
  const int arow  = (wid & 3) * 16 + l15;        // A tile row within block tile
  const int akb   = (lane >> 4) * 16;            // A lane K base (0 / 16)
  const int cpair = (wid >> 2) * 32;             // this wave's 32-col slice
  const int bcol0 = cpair + l15;
  const int bcol1 = bcol0 + 16;
  const int bkb   = (lane >> 4) * 16;            // B lane K base (0 / 16)

  v8f acc0 = {};
  v8f acc1 = {};

  for (int k0 = 0; k0 < K; k0 += 32) {
    __syncthreads();
    {
      const float* ap = A + (size_t)(rowBase + sr) * K + k0 + sq;
      float4 a0 = *(const float4*)(ap);
      float4 a1 = *(const float4*)(ap + 16);
      _Float16* d = &As[sr * 40 + sq];
      d[0]  = (_Float16)a0.x; d[1]  = (_Float16)a0.y; d[2]  = (_Float16)a0.z; d[3]  = (_Float16)a0.w;
      d[16] = (_Float16)a1.x; d[17] = (_Float16)a1.y; d[18] = (_Float16)a1.z; d[19] = (_Float16)a1.w;

      const float* wp = W + (size_t)(colBase + sr) * K + k0 + sq;
      float4 w0 = *(const float4*)(wp);
      float4 w1 = *(const float4*)(wp + 16);
      _Float16* e = &Bs[sr * 40 + sq];
      e[0]  = (_Float16)w0.x; e[1]  = (_Float16)w0.y; e[2]  = (_Float16)w0.z; e[3]  = (_Float16)w0.w;
      e[16] = (_Float16)w1.x; e[17] = (_Float16)w1.y; e[18] = (_Float16)w1.z; e[19] = (_Float16)w1.w;

      if (k0 + 32 < K) {                       // CDNA5 global_prefetch_b8 path
        __builtin_prefetch(ap + 32, 0, 0);
        __builtin_prefetch(wp + 32, 0, 0);
      }
    }
    __syncthreads();

    FragA fa;
    fa.q[0] = *(const uint4*)&As[arow * 40 + akb];        // K = kb..kb+7
    fa.q[1] = *(const uint4*)&As[arow * 40 + akb + 8];    // K = kb+8..kb+15
    FragA fb0;
    fb0.q[0] = *(const uint4*)&Bs[bcol0 * 40 + bkb];      // K = kb2..kb2+7
    fb0.q[1] = *(const uint4*)&Bs[bcol0 * 40 + bkb + 8];  // K = kb2+8..kb2+15
    FragA fb1;
    fb1.q[0] = *(const uint4*)&Bs[bcol1 * 40 + bkb];
    fb1.q[1] = *(const uint4*)&Bs[bcol1 * 40 + bkb + 8];

    acc0 = wmma_f16(fa.h, fb0.h, acc0);
    acc1 = wmma_f16(fa.h, fb1.h, acc1);
  }

  // Epilogue: C layout — lane = column, VGPR g = row g (+8 for upper half-wave)
  const int col0 = colBase + bcol0;
  const int col1 = colBase + bcol1;
  const float b0 = bvec[col0];
  const float b1 = bvec[col1];
  const int rb = rowBase + (wid & 3) * 16 + (lane >> 4) * 8;
#pragma unroll
  for (int g = 0; g < 8; ++g) {
    float v0 = acc0[g] + b0;
    float v1 = acc1[g] + b1;
    if (ACT == 1) { v0 = gelu_exact(v0); v1 = gelu_exact(v1); }
    C[(size_t)(rb + g) * Nout + col0] = v0;
    C[(size_t)(rb + g) * Nout + col1] = v1;
  }
}

// ---------------------------------------------------------------------------
// Attention: one wave per (b, h, 16-query tile). Flash-style streaming over
// 32-key chunks with fp32 online softmax; QK^T and PV through WMMA.
// qkv layout: [B, N, 3*D] (Q | K | V), bias: [B, N, N], out: [B, N, D].
// ---------------------------------------------------------------------------
__global__ __launch_bounds__(32) void attn_kernel(const float* __restrict__ qkv,
                                                  const float* __restrict__ bias,
                                                  float* __restrict__ out) {
  __shared__ __align__(16) _Float16 p_lds[16 * 40];   // P tile (16 q x 32 keys)
  __shared__ __align__(16) _Float16 vt[32 * 40];      // V^T tile (32 hd x 32 keys)

  const int lane  = threadIdx.x;
  const int bh    = blockIdx.y;
  const int b     = bh >> 3;
  const int h     = bh & 7;
  const int qbase = blockIdx.x * 16;

  const float* base = qkv + (size_t)b * SEQ * (3 * D_MODEL);
  const float* qp = base + h * HEADD;
  const float* kp = base + D_MODEL + h * HEADD;
  const float* vp = base + 2 * D_MODEL + h * HEADD;
  const float* bb = bias + (size_t)b * SEQ * SEQ;

  const int l15   = lane & 15;
  const int half8 = (lane >> 4) * 8;
  const int bkb   = (lane >> 4) * 16;
  const float scale = 0.17677669529663687f;   // 1/sqrt(32), folded into Q

  // Q fragment (A layout: row = l15, halves = hd dims {kb..kb+15})
  FragA fq;
  {
    const float* qr = qp + (size_t)(qbase + l15) * (3 * D_MODEL) + bkb;
    float4 a0 = *(const float4*)(qr);
    float4 a1 = *(const float4*)(qr + 4);
    float4 a2 = *(const float4*)(qr + 8);
    float4 a3 = *(const float4*)(qr + 12);
    fq.e[0]  = (_Float16)(a0.x * scale); fq.e[1]  = (_Float16)(a0.y * scale);
    fq.e[2]  = (_Float16)(a0.z * scale); fq.e[3]  = (_Float16)(a0.w * scale);
    fq.e[4]  = (_Float16)(a1.x * scale); fq.e[5]  = (_Float16)(a1.y * scale);
    fq.e[6]  = (_Float16)(a1.z * scale); fq.e[7]  = (_Float16)(a1.w * scale);
    fq.e[8]  = (_Float16)(a2.x * scale); fq.e[9]  = (_Float16)(a2.y * scale);
    fq.e[10] = (_Float16)(a2.z * scale); fq.e[11] = (_Float16)(a2.w * scale);
    fq.e[12] = (_Float16)(a3.x * scale); fq.e[13] = (_Float16)(a3.y * scale);
    fq.e[14] = (_Float16)(a3.z * scale); fq.e[15] = (_Float16)(a3.w * scale);
  }

  float mrow[8], lrow[8];
#pragma unroll
  for (int g = 0; g < 8; ++g) { mrow[g] = -INFINITY; lrow[g] = 0.0f; }
  v8f o0 = {};
  v8f o1 = {};

  for (int n0 = 0; n0 < SEQ; n0 += 32) {
    // K fragments directly from global (B layout: col = key, halves = hd dims)
    FragA fk0, fk1;
    load16h(fk0, kp + (size_t)(n0 + l15) * (3 * D_MODEL) + bkb);
    load16h(fk1, kp + (size_t)(n0 + 16 + l15) * (3 * D_MODEL) + bkb);

    v8f z = {};
    v8f s0 = wmma_f16(fq.h, fk0.h, z);   // rows=queries, cols=keys n0..n0+15
    v8f s1 = wmma_f16(fq.h, fk1.h, z);   // cols=keys n0+16..n0+31

    // Stage V^T into LDS (each lane owns one key row)
    {
      const float* vr = vp + (size_t)(n0 + lane) * (3 * D_MODEL);
#pragma unroll
      for (int d4 = 0; d4 < HEADD; d4 += 4) {
        float4 vv = *(const float4*)(vr + d4);
        vt[(d4 + 0) * 40 + lane] = (_Float16)vv.x;
        vt[(d4 + 1) * 40 + lane] = (_Float16)vv.y;
        vt[(d4 + 2) * 40 + lane] = (_Float16)vv.z;
        vt[(d4 + 3) * 40 + lane] = (_Float16)vv.w;
      }
    }

    const int key0 = n0 + l15;
#pragma unroll
    for (int g = 0; g < 8; ++g) {
      const size_t boff = (size_t)(qbase + half8 + g) * SEQ;
      float s0v = s0[g] + bb[boff + key0];
      float s1v = s1[g] + bb[boff + key0 + 16];

      float t = fmaxf(s0v, s1v);
      t = fmaxf(t, __shfl_xor(t, 1, 16));
      t = fmaxf(t, __shfl_xor(t, 2, 16));
      t = fmaxf(t, __shfl_xor(t, 4, 16));
      t = fmaxf(t, __shfl_xor(t, 8, 16));

      float mn = fmaxf(mrow[g], t);
      float p0 = 0.0f, p1 = 0.0f, corr = 1.0f;
      if (mn != -INFINITY) {
        corr = __expf(mrow[g] - mn);   // mrow == -inf -> 0
        p0 = __expf(s0v - mn);         // s == -inf -> 0
        p1 = __expf(s1v - mn);
      }
      float ps = p0 + p1;
      ps += __shfl_xor(ps, 1, 16);
      ps += __shfl_xor(ps, 2, 16);
      ps += __shfl_xor(ps, 4, 16);
      ps += __shfl_xor(ps, 8, 16);

      lrow[g] = lrow[g] * corr + ps;
      mrow[g] = mn;
      o0[g] *= corr;
      o1[g] *= corr;

      const int mrw = half8 + g;
      p_lds[mrw * 40 + l15] = (_Float16)p0;
      p_lds[mrw * 40 + 16 + l15] = (_Float16)p1;
    }
    __syncthreads();

    FragA fp;     // A layout: row = query, K = 32 keys of this chunk
    fp.q[0] = *(const uint4*)&p_lds[l15 * 40 + bkb];
    fp.q[1] = *(const uint4*)&p_lds[l15 * 40 + bkb + 8];
    FragA fv0;    // B layout: col = hd dim 0..15, K = keys
    fv0.q[0] = *(const uint4*)&vt[l15 * 40 + bkb];
    fv0.q[1] = *(const uint4*)&vt[l15 * 40 + bkb + 8];
    FragA fv1;    // hd dims 16..31
    fv1.q[0] = *(const uint4*)&vt[(16 + l15) * 40 + bkb];
    fv1.q[1] = *(const uint4*)&vt[(16 + l15) * 40 + bkb + 8];

    o0 = wmma_f16(fp.h, fv0.h, o0);
    o1 = wmma_f16(fp.h, fv1.h, o1);
    __syncthreads();
  }

  float* op = out + (size_t)b * SEQ * D_MODEL + h * HEADD;
#pragma unroll
  for (int g = 0; g < 8; ++g) {
    float inv = (lrow[g] > 0.0f) ? 1.0f / lrow[g] : 0.0f;
    const size_t r = (size_t)(qbase + half8 + g) * D_MODEL;
    op[r + l15] = o0[g] * inv;
    op[r + 16 + l15] = o1[g] * inv;
  }
}

// ---------------------------------------------------------------------------
// Fused residual + LayerNorm over D=256. One 256-thread block per row.
// res may be nullptr (plain LN). Safe for in-place x == y.
// ---------------------------------------------------------------------------
__global__ __launch_bounds__(256) void ln_kernel(const float* __restrict__ x,
                                                 const float* __restrict__ res,
                                                 const float* __restrict__ w,
                                                 const float* __restrict__ bvec,
                                                 float* __restrict__ y) {
  __shared__ float red[256];
  __shared__ float s_mu, s_rstd;
  const int row = blockIdx.x;
  const int i = threadIdx.x;
  float v = x[(size_t)row * D_MODEL + i];
  if (res) v += res[(size_t)row * D_MODEL + i];

  red[i] = v;
  __syncthreads();
  for (int off = 128; off; off >>= 1) {
    if (i < off) red[i] += red[i + off];
    __syncthreads();
  }
  if (i == 0) s_mu = red[0] * (1.0f / D_MODEL);
  __syncthreads();

  float d = v - s_mu;
  red[i] = d * d;
  __syncthreads();
  for (int off = 128; off; off >>= 1) {
    if (i < off) red[i] += red[i + off];
    __syncthreads();
  }
  if (i == 0) s_rstd = rsqrtf(red[0] * (1.0f / D_MODEL) + 1e-5f);
  __syncthreads();

  y[(size_t)row * D_MODEL + i] = d * s_rstd * w[i] + bvec[i];
}

// ---------------------------------------------------------------------------
// Host orchestration
// ---------------------------------------------------------------------------
extern "C" void kernel_launch(void* const* d_in, const int* in_sizes, int n_in,
                              void* d_out, int out_size, void* d_ws, size_t ws_size,
                              hipStream_t stream) {
  (void)in_sizes; (void)n_in; (void)out_size; (void)ws_size;

  const float* H     = (const float*)d_in[0];
  const int*   ei    = (const int*)d_in[1];
  const float* m     = (const float*)d_in[2];
  const float* Wqkv  = (const float*)d_in[3];
  const float* bqkv  = (const float*)d_in[4];
  const float* Wo    = (const float*)d_in[5];
  const float* bo    = (const float*)d_in[6];
  const float* ln1w  = (const float*)d_in[7];
  const float* ln1b  = (const float*)d_in[8];
  const float* W1    = (const float*)d_in[9];
  const float* b1    = (const float*)d_in[10];
  const float* W2    = (const float*)d_in[11];
  const float* b2    = (const float*)d_in[12];
  const float* ln2w  = (const float*)d_in[13];
  const float* ln2b  = (const float*)d_in[14];
  const float* lnfw  = (const float*)d_in[15];
  const float* lnfb  = (const float*)d_in[16];

  float* ws = (float*)d_ws;
  const size_t NN    = (size_t)SEQ * SEQ;
  const size_t ROWSZ = (size_t)BATCH * SEQ * D_MODEL;     // 2M floats
  float* bias = ws;                                       // 8M floats (32 MB)
  float* xbuf = bias + (size_t)BATCH * NN;                // 2M
  float* qkv  = xbuf + ROWSZ;                             // 6M
  float* attn = qkv + (size_t)BATCH * SEQ * 3 * D_MODEL;  // 2M
  float* proj = attn + ROWSZ;                             // 2M (also ffn output)
  float* ffh  = qkv;  // reuse qkv+attn region (8M floats) for FFN hidden

  // Bias construction
  bias_init_kernel<<<(unsigned)((BATCH * NN) / 256), 256, 0, stream>>>(bias);
  edge_scatter_kernel<<<(BATCH * NEDGE) / 256, 256, 0, stream>>>(ei, m, bias);

  const int Mrows = BATCH * SEQ;   // 8192
  for (int l = 0; l < NLAYER; ++l) {
    const float* xin = l ? (const float*)xbuf : H;
    // QKV projection: [8192,256] x [768,256]^T
    gemm_kernel<0><<<dim3(768 / 64, Mrows / 64), 256, 0, stream>>>(
        xin, Wqkv + (size_t)l * 3 * D_MODEL * D_MODEL, bqkv + (size_t)l * 3 * D_MODEL,
        qkv, Mrows, D_MODEL, 3 * D_MODEL);
    // Attention with bias
    attn_kernel<<<dim3(SEQ / 16, BATCH * NHEAD), 32, 0, stream>>>(qkv, bias, attn);
    // Output projection
    gemm_kernel<0><<<dim3(D_MODEL / 64, Mrows / 64), 256, 0, stream>>>(
        attn, Wo + (size_t)l * D_MODEL * D_MODEL, bo + (size_t)l * D_MODEL,
        proj, Mrows, D_MODEL, D_MODEL);
    // x = LN1(x + attn_proj)
    ln_kernel<<<Mrows, 256, 0, stream>>>(xin, proj, ln1w + (size_t)l * D_MODEL,
                                         ln1b + (size_t)l * D_MODEL, xbuf);
    // FFN up + GELU: [8192,256] x [1024,256]^T
    gemm_kernel<1><<<dim3(FFDIM / 64, Mrows / 64), 256, 0, stream>>>(
        xbuf, W1 + (size_t)l * FFDIM * D_MODEL, b1 + (size_t)l * FFDIM,
        ffh, Mrows, D_MODEL, FFDIM);
    // FFN down: [8192,1024] x [256,1024]^T
    gemm_kernel<0><<<dim3(D_MODEL / 64, Mrows / 64), 256, 0, stream>>>(
        ffh, W2 + (size_t)l * D_MODEL * FFDIM, b2 + (size_t)l * D_MODEL,
        proj, Mrows, FFDIM, D_MODEL);
    // x = LN2(x + ffn)
    ln_kernel<<<Mrows, 256, 0, stream>>>(xbuf, proj, ln2w + (size_t)l * D_MODEL,
                                         ln2b + (size_t)l * D_MODEL, xbuf);
  }
  // Final LN -> output
  ln_kernel<<<Mrows, 256, 0, stream>>>(xbuf, nullptr, lnfw, lnfb, (float*)d_out);
}
